// Encoder_65773129171109
// MI455X (gfx1250) — matrix operand order repaired and checked by reading, not compiled, not run on hardware
//
#include <hip/hip_runtime.h>

#define EMB 384
#define H2  768
#define NNODES 50000
#define NEDGES 400000
#define NGRAPHS 128
#define NLAYERS 5
#define BN_EPS 1e-5f

typedef __attribute__((ext_vector_type(16))) __bf16 v16bf;
typedef __attribute__((ext_vector_type(8)))  __bf16 v8bf;
typedef __attribute__((ext_vector_type(4)))  __bf16 v4bf;
typedef __attribute__((ext_vector_type(8)))  float  v8f;

__device__ __forceinline__ v16bf cat8(v8bf a, v8bf b) {
  return __builtin_shufflevector(a, b, 0, 1, 2, 3, 4, 5, 6, 7,
                                 8, 9, 10, 11, 12, 13, 14, 15);
}

// ---------------------------------------------------------------------------
// One-time weight prep: transpose to [l][n][k] and split into bf16 hi+lo so
// GEMM B-fragments are contiguous 32-byte runs in both global and LDS.
// ---------------------------------------------------------------------------
__global__ void split_wT_kernel(const float* __restrict__ W,
                                __bf16* __restrict__ hiT,
                                __bf16* __restrict__ loT,
                                int Lcount, int K, int Ncol) {
  size_t total = (size_t)Lcount * K * Ncol;
  size_t i = (size_t)blockIdx.x * blockDim.x + threadIdx.x;
  size_t st = (size_t)gridDim.x * blockDim.x;
  for (; i < total; i += st) {
    size_t k = i % (size_t)K;
    size_t t = i / (size_t)K;
    size_t n = t % (size_t)Ncol;
    size_t l = t / (size_t)Ncol;
    float f = W[(l * K + k) * Ncol + n];
    __bf16 h = (__bf16)f;
    hiT[i] = h;
    loT[i] = (__bf16)(f - (float)h);
  }
}

__global__ void zero_f32_kernel(float* __restrict__ p, size_t n) {
  size_t i = (size_t)blockIdx.x * blockDim.x + threadIdx.x;
  size_t st = (size_t)gridDim.x * blockDim.x;
  for (; i < n; i += st) p[i] = 0.0f;
}

// ---------------------------------------------------------------------------
// AtomEncoder: h[n,c] = sum_col atom_emb[col, x[n,col], c]
// ---------------------------------------------------------------------------
__global__ __launch_bounds__(EMB)
void encode_nodes_kernel(const int* __restrict__ x,
                         const float* __restrict__ atom_emb,
                         float* __restrict__ h) {
  int n = blockIdx.x;
  int c = threadIdx.x;
  float s = 0.0f;
#pragma unroll
  for (int col = 0; col < 9; ++col) {
    int idx = x[n * 9 + col];
    s += atom_emb[((size_t)(col * 64 + idx)) * EMB + c];
  }
  h[(size_t)n * EMB + c] = s;
}

// agg = h (vectorized copy; z = h + agg folds the self term in for free)
__global__ void copy_f4_kernel(const float* __restrict__ src,
                               float* __restrict__ dst, size_t n4) {
  size_t i = (size_t)blockIdx.x * blockDim.x + threadIdx.x;
  size_t st = (size_t)gridDim.x * blockDim.x;
  const float4* s = (const float4*)src;
  float4* d = (float4*)dst;
  for (; i < n4; i += st) d[i] = s[i];
}

// ---------------------------------------------------------------------------
// GINE message + scatter: agg[dst] += relu(h[src] + bond_embed(edge_attr))
// bond table is 36KB -> L2 resident, recomputed per layer instead of a
// 614MB materialized e-buffer.
// ---------------------------------------------------------------------------
__global__ __launch_bounds__(EMB)
void scatter_edges_kernel(const int* __restrict__ edge_index,
                          const int* __restrict__ edge_attr,
                          const float* __restrict__ bond_emb,
                          const float* __restrict__ h,
                          float* __restrict__ agg) {
  int e = blockIdx.x;
  int c = threadIdx.x;
  int src = edge_index[e];
  int dst = edge_index[NEDGES + e];
  int a0 = edge_attr[e * 3 + 0];
  int a1 = edge_attr[e * 3 + 1];
  int a2 = edge_attr[e * 3 + 2];
  float ev = bond_emb[(size_t)(0 * 8 + a0) * EMB + c] +
             bond_emb[(size_t)(1 * 8 + a1) * EMB + c] +
             bond_emb[(size_t)(2 * 8 + a2) * EMB + c];
  float m = h[(size_t)src * EMB + c] + ev;
  m = m > 0.0f ? m : 0.0f;
  __hip_atomic_fetch_add(&agg[(size_t)dst * EMB + c], m,
                         __ATOMIC_RELAXED, __HIP_MEMORY_SCOPE_AGENT);
}

// ---------------------------------------------------------------------------
// bf16x2 split GEMM on v_wmma_f32_16x16x32_bf16 (wave32).
// Out[N x Ncol] = A[N x K] @ W[K x Ncol] + bias, fp32 in/out,
// C += Ahi*Whi + Ahi*Wlo + Alo*Whi  (error ~2^-16 relative).
// Tiles staged in LDS *in fragment order* (A rows permuted by swapping bits
// 3<->4 of k; W pre-transposed), so every fragment is two ds_load_b128.
// Stride 40 halves (80B) keeps 16B alignment and conflict-free banks.
// ---------------------------------------------------------------------------
#define TBM 128
#define TBN 64
#define TBK 32
#define ASTR 40
#define WSTR 40

__global__ __launch_bounds__(256)
void gemm_bf16x2_kernel(const float* __restrict__ A,
                        const __bf16* __restrict__ WhiT,
                        const __bf16* __restrict__ WloT,
                        const float* __restrict__ bias,
                        float* __restrict__ Out,
                        int Nrows, int K, int Ncol) {
  __shared__ __align__(16) __bf16 sAh[TBM * ASTR];  // 10 KB
  __shared__ __align__(16) __bf16 sAl[TBM * ASTR];  // 10 KB
  __shared__ __align__(16) __bf16 sWh[TBN * WSTR];  // 5 KB
  __shared__ __align__(16) __bf16 sWl[TBN * WSTR];  // 5 KB

  const int tid  = threadIdx.x;
  const int wave = tid >> 5;
  const int lane = tid & 31;
  const int lrow = lane & 15;
  const int lhi  = lane >> 4;

  const int rowBlock = blockIdx.x * TBM;
  const int colBlock = blockIdx.y * TBN;
  const int waveRow  = rowBlock + wave * 16;
  const bool fullTile = (rowBlock + TBM) <= Nrows;   // uniform per block

  v8f acc[4];
#pragma unroll
  for (int j = 0; j < 4; ++j) {
    v8f z = {0.f, 0.f, 0.f, 0.f, 0.f, 0.f, 0.f, 0.f};
    acc[j] = z;
  }

  for (int k0 = 0; k0 < K; k0 += TBK) {
    // ---- Stage A: split fp32 -> bf16 hi/lo, store rows permuted so a
    //      lane's fragment is one contiguous 32B run.
    if (fullTile) {
#pragma unroll
      for (int t = tid; t < TBM * 8; t += 256) {
        int r  = t >> 3;
        int k4 = (t & 7) * 4;
        const float* gp = A + (size_t)(rowBlock + r) * K + k0 + k4;
        float4 v = *(const float4*)gp;
        if (k0 + TBK < K) __builtin_prefetch(gp + TBK, 0, 3);  // WGP-scope
        int pos = (k4 & 7) | ((k4 >> 3 & 1) << 4) | ((k4 >> 4 & 1) << 3);
        v4bf ph, pl;
        __bf16 h0 = (__bf16)v.x; ph[0] = h0; pl[0] = (__bf16)(v.x - (float)h0);
        __bf16 h1 = (__bf16)v.y; ph[1] = h1; pl[1] = (__bf16)(v.y - (float)h1);
        __bf16 h2 = (__bf16)v.z; ph[2] = h2; pl[2] = (__bf16)(v.z - (float)h2);
        __bf16 h3 = (__bf16)v.w; ph[3] = h3; pl[3] = (__bf16)(v.w - (float)h3);
        *(v4bf*)(sAh + r * ASTR + pos) = ph;
        *(v4bf*)(sAl + r * ASTR + pos) = pl;
      }
    } else {
#pragma unroll
      for (int t = tid; t < TBM * 8; t += 256) {
        int r  = t >> 3;
        int k4 = (t & 7) * 4;
        int grow = rowBlock + r;
        float4 v = make_float4(0.f, 0.f, 0.f, 0.f);
        if (grow < Nrows) v = *(const float4*)(A + (size_t)grow * K + k0 + k4);
        int pos = (k4 & 7) | ((k4 >> 3 & 1) << 4) | ((k4 >> 4 & 1) << 3);
        v4bf ph, pl;
        __bf16 h0 = (__bf16)v.x; ph[0] = h0; pl[0] = (__bf16)(v.x - (float)h0);
        __bf16 h1 = (__bf16)v.y; ph[1] = h1; pl[1] = (__bf16)(v.y - (float)h1);
        __bf16 h2 = (__bf16)v.z; ph[2] = h2; pl[2] = (__bf16)(v.z - (float)h2);
        __bf16 h3 = (__bf16)v.w; ph[3] = h3; pl[3] = (__bf16)(v.w - (float)h3);
        *(v4bf*)(sAh + r * ASTR + pos) = ph;
        *(v4bf*)(sAl + r * ASTR + pos) = pl;
      }
    }
    // ---- Stage W (already transposed + split): pure 16B vector copies.
    {
      int arr  = tid >> 7;      // 0 = hi, 1 = lo
      int u    = tid & 127;
      int n    = u >> 1;
      int half = u & 1;
      const __bf16* src = (arr ? WloT : WhiT) +
                          (size_t)(colBlock + n) * K + k0 + half * 16;
      __bf16* dst = (arr ? sWl : sWh) + n * WSTR + half * 8;
      *(v8bf*)(dst)     = *(const v8bf*)(src);
      *(v8bf*)(dst + 16) = *(const v8bf*)(src + 8);
    }
    __syncthreads();

    // ---- A fragments: two ds_load_b128 each.
    const __bf16* ap  = sAh + (wave * 16 + lrow) * ASTR + lhi * 16;
    const __bf16* alp = sAl + (wave * 16 + lrow) * ASTR + lhi * 16;
    v16bf ah = cat8(*(const v8bf*)ap,  *(const v8bf*)(ap + 8));
    v16bf al = cat8(*(const v8bf*)alp, *(const v8bf*)(alp + 8));

#pragma unroll
    for (int j = 0; j < 4; ++j) {
      const __bf16* bp  = sWh + (j * 16 + lrow) * WSTR + lhi * 16;
      const __bf16* blp = sWl + (j * 16 + lrow) * WSTR + lhi * 16;
      v16bf bh = cat8(*(const v8bf*)bp,  *(const v8bf*)(bp + 8));
      v16bf bl = cat8(*(const v8bf*)blp, *(const v8bf*)(blp + 8));
      acc[j] = __builtin_amdgcn_wmma_f32_16x16x32_bf16(
          false, ah, false, bh, (short)0, acc[j], false, false);
      acc[j] = __builtin_amdgcn_wmma_f32_16x16x32_bf16(
          false, ah, false, bl, (short)0, acc[j], false, false);
      acc[j] = __builtin_amdgcn_wmma_f32_16x16x32_bf16(
          false, al, false, bh, (short)0, acc[j], false, false);
    }
    __syncthreads();
  }

  // C/D layout: VGPR r -> row = r + 8*lhi, col = lrow within each 16x16 tile
  if (fullTile) {
#pragma unroll
    for (int j = 0; j < 4; ++j) {
      int gcol = colBlock + j * 16 + lrow;
      float b = bias[gcol];
#pragma unroll
      for (int r = 0; r < 8; ++r) {
        int grow = waveRow + r + lhi * 8;
        Out[(size_t)grow * Ncol + gcol] = acc[j][r] + b;
      }
    }
  } else {
#pragma unroll
    for (int j = 0; j < 4; ++j) {
      int gcol = colBlock + j * 16 + lrow;
      float b = bias[gcol];
#pragma unroll
      for (int r = 0; r < 8; ++r) {
        int grow = waveRow + r + lhi * 8;
        if (grow < Nrows) Out[(size_t)grow * Ncol + gcol] = acc[j][r] + b;
      }
    }
  }
}

// ---------------------------------------------------------------------------
// Training-mode BatchNorm over the node dim: column sums / sum-of-squares
// ---------------------------------------------------------------------------
__global__ void col_stats_kernel(const float* __restrict__ z,
                                 float* __restrict__ sum,
                                 float* __restrict__ sumsq,
                                 int Nrows, int C, int rowsPerBlock) {
  int c = threadIdx.x;
  int r0 = blockIdx.x * rowsPerBlock;
  int r1 = r0 + rowsPerBlock;
  if (r1 > Nrows) r1 = Nrows;
  float s = 0.f, q = 0.f;
  for (int r = r0; r < r1; ++r) {
    float v = z[(size_t)r * C + c];
    s += v;
    q += v * v;
  }
  __hip_atomic_fetch_add(&sum[c], s, __ATOMIC_RELAXED, __HIP_MEMORY_SCOPE_AGENT);
  __hip_atomic_fetch_add(&sumsq[c], q, __ATOMIC_RELAXED, __HIP_MEMORY_SCOPE_AGENT);
}

__global__ void finalize_stats_kernel(const float* __restrict__ sum,
                                      const float* __restrict__ sumsq,
                                      const float* __restrict__ gamma,
                                      const float* __restrict__ beta,
                                      float* __restrict__ scale,
                                      float* __restrict__ shift,
                                      int C, float invN) {
  int c = blockIdx.x * blockDim.x + threadIdx.x;
  if (c >= C) return;
  float m = sum[c] * invN;
  float v = sumsq[c] * invN - m * m;
  float sc = gamma[c] * rsqrtf(v + BN_EPS);
  scale[c] = sc;
  shift[c] = beta[c] - m * sc;
}

__global__ void bn_relu_kernel(const float* __restrict__ z,
                               const float* __restrict__ scale,
                               const float* __restrict__ shift,
                               float* __restrict__ out, int C) {
  int n = blockIdx.x;
  int c = threadIdx.x;
  float v = z[(size_t)n * C + c] * scale[c] + shift[c];
  out[(size_t)n * C + c] = v > 0.f ? v : 0.f;
}

// global_add_pool
__global__ __launch_bounds__(EMB)
void pool_kernel(const float* __restrict__ h, const int* __restrict__ batch,
                 float* __restrict__ gout) {
  int n = blockIdx.x;
  int c = threadIdx.x;
  int g = batch[n];
  __hip_atomic_fetch_add(&gout[(size_t)g * EMB + c], h[(size_t)n * EMB + c],
                         __ATOMIC_RELAXED, __HIP_MEMORY_SCOPE_AGENT);
}

// ---------------------------------------------------------------------------
extern "C" void kernel_launch(void* const* d_in, const int* in_sizes, int n_in,
                              void* d_out, int out_size, void* d_ws, size_t ws_size,
                              hipStream_t stream) {
  (void)in_sizes; (void)n_in; (void)out_size; (void)ws_size;

  const int*   batch     = (const int*)d_in[0];
  const int*   x         = (const int*)d_in[1];
  const int*   edge_idx  = (const int*)d_in[2];
  const int*   edge_attr = (const int*)d_in[3];
  const float* atom_emb  = (const float*)d_in[4];
  const float* bond_emb  = (const float*)d_in[5];
  const float* W1        = (const float*)d_in[6];
  const float* b1        = (const float*)d_in[7];
  const float* g1        = (const float*)d_in[8];
  const float* be1       = (const float*)d_in[9];
  const float* W2        = (const float*)d_in[10];
  const float* b2        = (const float*)d_in[11];
  const float* gbn       = (const float*)d_in[12];
  const float* bbn       = (const float*)d_in[13];

  float* gout = (float*)d_out;                  // [128, 384]
  float* hbuf = gout + (size_t)NGRAPHS * EMB;   // [N, 384] lives in d_out

  // Workspace carve-up
  char* ws = (char*)d_ws;
  size_t off = 0;
  auto take = [&](size_t bytes) -> char* {
    char* p = ws + off;
    off = (off + bytes + 255) & ~(size_t)255;
    return p;
  };
  float*  agg   = (float*)take((size_t)NNODES * EMB * 4);
  float*  z1    = (float*)take((size_t)NNODES * H2 * 4);
  float*  z2    = (float*)take((size_t)NNODES * EMB * 4);
  __bf16* W1hiT = (__bf16*)take((size_t)NLAYERS * EMB * H2 * 2);
  __bf16* W1loT = (__bf16*)take((size_t)NLAYERS * EMB * H2 * 2);
  __bf16* W2hiT = (__bf16*)take((size_t)NLAYERS * H2 * EMB * 2);
  __bf16* W2loT = (__bf16*)take((size_t)NLAYERS * H2 * EMB * 2);
  float*  ssum  = (float*)take(H2 * 4);
  float*  ssq   = (float*)take(H2 * 4);
  float*  scale = (float*)take(H2 * 4);
  float*  shift = (float*)take(H2 * 4);

  // Precision prep: transpose + split both weight stacks into bf16 hi/lo
  split_wT_kernel<<<1024, 256, 0, stream>>>(W1, W1hiT, W1loT, NLAYERS, EMB, H2);
  split_wT_kernel<<<1024, 256, 0, stream>>>(W2, W2hiT, W2loT, NLAYERS, H2, EMB);

  // Atom encoder
  encode_nodes_kernel<<<NNODES, EMB, 0, stream>>>(x, atom_emb, hbuf);

  const size_t nh4 = (size_t)NNODES * EMB / 4;

  for (int i = 0; i < NLAYERS; ++i) {
    // agg = h ; agg[dst] += relu(h[src] + e)
    copy_f4_kernel<<<4096, 256, 0, stream>>>(hbuf, agg, nh4);
    scatter_edges_kernel<<<NEDGES, EMB, 0, stream>>>(edge_idx, edge_attr,
                                                     bond_emb, hbuf, agg);

    // z1 = agg @ W1[i] + b1[i]
    {
      dim3 grid((NNODES + TBM - 1) / TBM, H2 / TBN);
      gemm_bf16x2_kernel<<<grid, 256, 0, stream>>>(
          agg, W1hiT + (size_t)i * EMB * H2, W1loT + (size_t)i * EMB * H2,
          b1 + (size_t)i * H2, z1, NNODES, EMB, H2);
    }
    // BN(768) + ReLU in place
    zero_f32_kernel<<<8, 256, 0, stream>>>(ssum, H2);
    zero_f32_kernel<<<8, 256, 0, stream>>>(ssq, H2);
    col_stats_kernel<<<(NNODES + 127) / 128, H2, 0, stream>>>(z1, ssum, ssq,
                                                              NNODES, H2, 128);
    finalize_stats_kernel<<<(H2 + 255) / 256, 256, 0, stream>>>(
        ssum, ssq, g1 + (size_t)i * H2, be1 + (size_t)i * H2,
        scale, shift, H2, 1.0f / NNODES);
    bn_relu_kernel<<<NNODES, H2, 0, stream>>>(z1, scale, shift, z1, H2);

    // z2 = z1 @ W2[i] + b2[i]
    {
      dim3 grid((NNODES + TBM - 1) / TBM, EMB / TBN);
      gemm_bf16x2_kernel<<<grid, 256, 0, stream>>>(
          z1, W2hiT + (size_t)i * H2 * EMB, W2loT + (size_t)i * H2 * EMB,
          b2 + (size_t)i * EMB, z2, NNODES, H2, EMB);
    }
    // BN(384) + ReLU -> h
    zero_f32_kernel<<<4, 256, 0, stream>>>(ssum, EMB);
    zero_f32_kernel<<<4, 256, 0, stream>>>(ssq, EMB);
    col_stats_kernel<<<(NNODES + 127) / 128, EMB, 0, stream>>>(z2, ssum, ssq,
                                                               NNODES, EMB, 128);
    finalize_stats_kernel<<<(EMB + 255) / 256, 256, 0, stream>>>(
        ssum, ssq, gbn + (size_t)i * EMB, bbn + (size_t)i * EMB,
        scale, shift, EMB, 1.0f / NNODES);
    bn_relu_kernel<<<NNODES, EMB, 0, stream>>>(z2, scale, shift, hbuf, EMB);
  }

  // global_add_pool into the first region of d_out
  zero_f32_kernel<<<64, 256, 0, stream>>>(gout, (size_t)NGRAPHS * EMB);
  pool_kernel<<<NNODES, EMB, 0, stream>>>(hbuf, batch, gout);
}
